// EllEssTeeEmm_16295105921440
// MI455X (gfx1250) — matrix operand order repaired and checked by reading, not compiled, not run on hardware
//
#include <hip/hip_runtime.h>

#define BB 32
#define LL 1024
#define DD 256
#define HH 512
#define G4 2048 // 4*H

typedef __attribute__((ext_vector_type(16))) __bf16 v16bf;
typedef __attribute__((ext_vector_type(8)))  float  v8f;
typedef unsigned int u32x4 __attribute__((ext_vector_type(4)));
typedef int          i32x4 __attribute__((ext_vector_type(4)));
typedef int          i32x8 __attribute__((ext_vector_type(8)));

union Frag {
    uint4 u4[2];
    v16bf v;
};

__device__ __forceinline__ unsigned short f2bf(float f) {
    unsigned u = __builtin_bit_cast(unsigned, f);
    u = (u + 0x7FFFu + ((u >> 16) & 1u)) >> 16;  // round-to-nearest-even
    return (unsigned short)u;
}
__device__ __forceinline__ float bf2f(unsigned short h) {
    unsigned u = ((unsigned)h) << 16;
    return __builtin_bit_cast(float, u);
}
__device__ __forceinline__ float sigmoidf_(float x) {
    return 1.0f / (1.0f + __expf(-x));
}
// stable fast tanh: 2 transcendentals -> v_exp_f32, no libm polynomial
__device__ __forceinline__ float tanhf_(float x) {
    float t = __expf(-2.0f * fabsf(x));
    float r = (1.0f - t) / (1.0f + t);
    return copysignf(r, x);
}

// ---------------- elementwise helpers ----------------
__global__ void k_cvt_bf16(const float* __restrict__ in, unsigned short* __restrict__ out, size_t n) {
    size_t i = (size_t)blockIdx.x * blockDim.x + threadIdx.x;
    size_t stride = (size_t)gridDim.x * blockDim.x;
    for (; i < n; i += stride) out[i] = f2bf(in[i]);
}
__global__ void k_bias(const float* __restrict__ a, const float* __restrict__ b,
                       float* __restrict__ o, int n) {
    int i = blockIdx.x * blockDim.x + threadIdx.x;
    if (i < n) o[i] = a[i] + b[i];
}
__global__ void k_zero_u32(unsigned* p) {
    if (threadIdx.x == 0 && blockIdx.x == 0) *p = 0u;
}

// ---------------- input-projection GEMM:  pre = A @ W^T + bias ----------------
// A: [32768, K] bf16 row-major (x or h01).  W: [2048, K] bf16 row-major.
// Output pre: time-major bf16 [L][B][2048].
// Each wave computes a 16x64 output strip (4 accumulators, shared A fragment).
__global__ __launch_bounds__(256)
void k_gemm_pre(const unsigned short* __restrict__ A,
                const unsigned short* __restrict__ W,
                const float* __restrict__ bias,
                unsigned short* __restrict__ pre,
                int K)
{
    const int lane = threadIdx.x & 31;
    const int wave = threadIdx.x >> 5;
    const int task = blockIdx.x * 8 + wave;   // 65536 wave-tasks
    const int mtile = task >> 5;              // 0..2047
    const int ngrp  = task & 31;              // 0..31 (each = 4 N-tiles)
    const int half  = lane >> 4;
    const int l16   = lane & 15;
    const int mrow  = mtile * 16 + l16;       // A row held by this lane

    v8f acc[4];
#pragma unroll
    for (int i = 0; i < 4; ++i) {
        const int n = (ngrp * 4 + i) * 16 + l16;
        const float b = bias[n];
#pragma unroll
        for (int r = 0; r < 8; ++r) acc[i][r] = b;
    }

    for (int kb = 0; kb < K; kb += 32) {
        Frag a;
        const unsigned short* ap = A + (size_t)mrow * K + kb + half * 8;
        a.u4[0] = *(const uint4*)(ap);
        a.u4[1] = *(const uint4*)(ap + 16);
#pragma unroll
        for (int i = 0; i < 4; ++i) {
            const int n = (ngrp * 4 + i) * 16 + l16;
            Frag bf;
            const unsigned short* wp = W + (size_t)n * K + kb + half * 8;
            bf.u4[0] = *(const uint4*)(wp);
            bf.u4[1] = *(const uint4*)(wp + 16);
            acc[i] = __builtin_amdgcn_wmma_f32_16x16x32_bf16(
                false, a.v, false, bf.v, (short)0, acc[i], false, false);
        }
    }

#pragma unroll
    for (int i = 0; i < 4; ++i) {
        const int n = (ngrp * 4 + i) * 16 + l16;
#pragma unroll
        for (int r = 0; r < 8; ++r) {
            const int mg = mtile * 16 + r + half * 8;   // global row = b*L + l
            const int b  = mg >> 10;
            const int l  = mg & (LL - 1);
            pre[((size_t)l * BB + b) * G4 + n] = f2bf(acc[i][r]);
        }
    }
}

// ---------------- persistent LSTM recurrence ----------------
// 8 workgroups x 256 threads.  WG bid owns hidden units [bid*64, bid*64+64).
// Wave v: M-tile = v>>2 (batch rows), N-subtile = v&3, covering the SAME 16
// hidden units across all four gate strips -> in-register c/h update.
__global__ __launch_bounds__(256)
void k_lstm_rec(const unsigned short* __restrict__ pre,   // [L][B][2048] bf16
                const unsigned short* __restrict__ Whh,   // [2048][512] bf16
                unsigned short* __restrict__ hstate,      // [2][B*H] bf16 ping-pong
                unsigned short* __restrict__ h01,         // [B][L][1024] bf16 or null
                float* __restrict__ fout,                 // [B][L][1024] f32 or null
                unsigned* __restrict__ cnt,
                int reverse, int dirOff)
{
    __shared__ __align__(16) unsigned short hsm[BB * HH];  // 32 KB prev-h stage

    const int lane = threadIdx.x & 31;
    const int wave = threadIdx.x >> 5;
    const int half = lane >> 4;
    const int l16  = lane & 15;
    const int mt   = wave >> 2;                       // 0..1
    const int nt   = wave & 3;                        // 0..3
    const int j    = blockIdx.x * 64 + nt * 16 + l16; // hidden unit column

    float c[8];
#pragma unroll
    for (int r = 0; r < 8; ++r) c[r] = 0.f;

    for (int s = 0; s < LL; ++s) {
        const int t  = reverse ? (LL - 1 - s) : s;
        const int rb = s & 1;
        const int wb = rb ^ 1;

        if (s > 0) {  // stage h_{t-1} into LDS
#if defined(__gfx1250__) && __has_builtin(__builtin_amdgcn_tensor_load_to_lds)
            // Tensor Data Mover: one 2D-tile DMA (512 x 32 rows of bf16,
            // row stride 512) global -> LDS, tracked with TENSORcnt.
            if (threadIdx.x < 32) {
                const unsigned long long ga =
                    (unsigned long long)(size_t)(hstate + (size_t)rb * (BB * HH));
                // LDS flat aperture truncates to addr[31:0] -> LDS byte offset
                const unsigned ldsb = (unsigned)(size_t)(void*)hsm;
                u32x4 g0;
                g0[0] = 1u;                                   // count=1, no gather
                g0[1] = ldsb;                                 // lds_addr
                g0[2] = (unsigned)(ga & 0xFFFFFFFFu);         // global_addr[31:0]
                g0[3] = (unsigned)((ga >> 32) & 0x01FFFFFFu)  // global_addr[56:32]
                        | (2u << 30);                         // type=2 (image/D#)
                i32x8 g1;
                g1[0] = (int)(1u << 16);            // data_size=1 (2 bytes)
                g1[1] = (int)(((unsigned)HH & 0xFFFFu) << 16); // tensor_dim0 lo16
                g1[2] = (int)(((unsigned)BB & 0xFFFFu) << 16); // dim0 hi=0 | tensor_dim1 lo16
                g1[3] = (int)(((unsigned)HH) << 16);           // dim1 hi=0 | tile_dim0
                g1[4] = (int)BB;                               // tile_dim1 | tile_dim2=0
                g1[5] = (int)HH;                               // tensor_dim0_stride[31:0]
                g1[6] = 0;                                     // stride hi / dim1_stride lo
                g1[7] = 0;
                i32x4 z4; z4[0] = z4[1] = z4[2] = z4[3] = 0;
#if __clang_major__ >= 23
                i32x8 z8; z8[0]=z8[1]=z8[2]=z8[3]=z8[4]=z8[5]=z8[6]=z8[7]=0;
                __builtin_amdgcn_tensor_load_to_lds(g0, g1, z4, z4, z8, 0);
#else
                __builtin_amdgcn_tensor_load_to_lds(g0, g1, z4, z4, 0);
#endif
#if __has_builtin(__builtin_amdgcn_s_wait_tensorcnt)
                __builtin_amdgcn_s_wait_tensorcnt(0);
#else
                asm volatile("s_wait_tensorcnt 0x0" ::: "memory");
#endif
            }
#else
            const uint4* src = (const uint4*)(hstate + (size_t)rb * (BB * HH));
            uint4* dst = (uint4*)hsm;
            for (int i = threadIdx.x; i < (BB * HH / 8); i += 256) dst[i] = src[i];
#endif
        }
        __syncthreads();

        // accumulators start from the precomputed input projection
        const unsigned short* ps = pre + (size_t)t * (BB * G4);
        v8f acc[4];
#pragma unroll
        for (int q = 0; q < 4; ++q) {
#pragma unroll
            for (int r = 0; r < 8; ++r) {
                const int m = mt * 16 + r + half * 8;
                acc[q][r] = bf2f(ps[(size_t)m * G4 + q * HH + j]);
            }
        }

        if (s > 0) {  // + h_{t-1} @ Whh^T  via WMMA, K=512 in 16 steps
            const int mrow = mt * 16 + l16;
            for (int kb = 0; kb < HH; kb += 32) {
                Frag a;
                const unsigned short* ap = hsm + mrow * HH + kb + half * 8;
                a.u4[0] = *(const uint4*)(ap);
                a.u4[1] = *(const uint4*)(ap + 16);
#pragma unroll
                for (int q = 0; q < 4; ++q) {
                    const int n = q * HH + j;
                    Frag bf;
                    const unsigned short* wp = Whh + (size_t)n * HH + kb + half * 8;
                    bf.u4[0] = *(const uint4*)(wp);
                    bf.u4[1] = *(const uint4*)(wp + 16);
                    acc[q] = __builtin_amdgcn_wmma_f32_16x16x32_bf16(
                        false, a.v, false, bf.v, (short)0, acc[q], false, false);
                }
            }
        }

        // prefetch next timestep's pre slice (streams from HBM/L2)
        if (s + 1 < LL) {
            const int tn = reverse ? (LL - 2 - s) : (s + 1);
            const unsigned short* pn = pre + (size_t)tn * (BB * G4);
            const int gtid = blockIdx.x * 256 + threadIdx.x;   // 0..2047
            __builtin_prefetch(pn + (size_t)gtid * 32, 0, 1);  // 64B / thread
        }

        // in-register gate math + state update + outputs
        unsigned short* hw = hstate + (size_t)wb * (BB * HH);
#pragma unroll
        for (int r = 0; r < 8; ++r) {
            const int m  = mt * 16 + r + half * 8;  // batch row
            const float ig = sigmoidf_(acc[0][r]);
            const float fg = sigmoidf_(acc[1][r]);
            const float gg = tanhf_(acc[2][r]);
            const float og = sigmoidf_(acc[3][r]);
            c[r] = fg * c[r] + ig * gg;
            const float h = og * tanhf_(c[r]);
            hw[(size_t)m * HH + j] = f2bf(h);
            const size_t oidx = ((size_t)m * LL + t) * (2 * HH) + dirOff + j;
            if (h01)  h01[oidx]  = f2bf(h);
            if (fout) fout[oidx] = h;
        }

        // global arrive/spin barrier across the 8 workgroups
        __syncthreads();
        if (threadIdx.x == 0) {
            __threadfence();
            atomicAdd(cnt, 1u);
            const unsigned target = 8u * (unsigned)(s + 1);
            while (__hip_atomic_load(cnt, __ATOMIC_ACQUIRE, __HIP_MEMORY_SCOPE_AGENT) < target) {
                __builtin_amdgcn_s_sleep(2);
            }
        }
        __syncthreads();
    }
}

// ---------------- host orchestration ----------------
extern "C" void kernel_launch(void* const* d_in, const int* in_sizes, int n_in,
                              void* d_out, int out_size, void* d_ws, size_t ws_size,
                              hipStream_t stream) {
    (void)in_sizes; (void)n_in; (void)out_size; (void)ws_size;
    const float* x      = (const float*)d_in[0];
    const float* wihf0  = (const float*)d_in[1];
    const float* whhf0  = (const float*)d_in[2];
    const float* bif0   = (const float*)d_in[3];
    const float* bhf0   = (const float*)d_in[4];
    const float* wihr0  = (const float*)d_in[5];
    const float* whhr0  = (const float*)d_in[6];
    const float* bir0   = (const float*)d_in[7];
    const float* bhr0   = (const float*)d_in[8];
    const float* wihf1  = (const float*)d_in[9];
    const float* whhf1  = (const float*)d_in[10];
    const float* bif1   = (const float*)d_in[11];
    const float* bhf1   = (const float*)d_in[12];
    const float* wihr1  = (const float*)d_in[13];
    const float* whhr1  = (const float*)d_in[14];
    const float* bir1   = (const float*)d_in[15];
    const float* bhr1   = (const float*)d_in[16];
    float* out = (float*)d_out;

    char* base = (char*)d_ws;
    size_t off = 0;
    auto alloc = [&](size_t bytes) { size_t o = off; off = (off + bytes + 255) & ~(size_t)255; return o; };

    unsigned short* xbf    = (unsigned short*)(base + alloc((size_t)BB * LL * DD * 2));
    unsigned short* bwihf0 = (unsigned short*)(base + alloc((size_t)G4 * DD * 2));
    unsigned short* bwihr0 = (unsigned short*)(base + alloc((size_t)G4 * DD * 2));
    unsigned short* bwihf1 = (unsigned short*)(base + alloc((size_t)G4 * 2 * HH * 2));
    unsigned short* bwihr1 = (unsigned short*)(base + alloc((size_t)G4 * 2 * HH * 2));
    unsigned short* bwhhf0 = (unsigned short*)(base + alloc((size_t)G4 * HH * 2));
    unsigned short* bwhhr0 = (unsigned short*)(base + alloc((size_t)G4 * HH * 2));
    unsigned short* bwhhf1 = (unsigned short*)(base + alloc((size_t)G4 * HH * 2));
    unsigned short* bwhhr1 = (unsigned short*)(base + alloc((size_t)G4 * HH * 2));
    float* bsf0 = (float*)(base + alloc(G4 * 4));
    float* bsr0 = (float*)(base + alloc(G4 * 4));
    float* bsf1 = (float*)(base + alloc(G4 * 4));
    float* bsr1 = (float*)(base + alloc(G4 * 4));
    unsigned short* pre0f = (unsigned short*)(base + alloc((size_t)LL * BB * G4 * 2));
    unsigned short* pre0r = (unsigned short*)(base + alloc((size_t)LL * BB * G4 * 2));
    unsigned short* pre1f = (unsigned short*)(base + alloc((size_t)LL * BB * G4 * 2));
    unsigned short* pre1r = (unsigned short*)(base + alloc((size_t)LL * BB * G4 * 2));
    unsigned short* h01   = (unsigned short*)(base + alloc((size_t)BB * LL * 2 * HH * 2));
    unsigned short* hst   = (unsigned short*)(base + alloc((size_t)2 * BB * HH * 2));
    unsigned*       cnt   = (unsigned*)(base + alloc(256));

    // 1) precision conversion
    k_cvt_bf16<<<4096, 256, 0, stream>>>(x, xbf, (size_t)BB * LL * DD);
    k_cvt_bf16<<<512, 256, 0, stream>>>(wihf0, bwihf0, (size_t)G4 * DD);
    k_cvt_bf16<<<512, 256, 0, stream>>>(wihr0, bwihr0, (size_t)G4 * DD);
    k_cvt_bf16<<<2048, 256, 0, stream>>>(wihf1, bwihf1, (size_t)G4 * 2 * HH);
    k_cvt_bf16<<<2048, 256, 0, stream>>>(wihr1, bwihr1, (size_t)G4 * 2 * HH);
    k_cvt_bf16<<<1024, 256, 0, stream>>>(whhf0, bwhhf0, (size_t)G4 * HH);
    k_cvt_bf16<<<1024, 256, 0, stream>>>(whhr0, bwhhr0, (size_t)G4 * HH);
    k_cvt_bf16<<<1024, 256, 0, stream>>>(whhf1, bwhhf1, (size_t)G4 * HH);
    k_cvt_bf16<<<1024, 256, 0, stream>>>(whhr1, bwhhr1, (size_t)G4 * HH);
    k_bias<<<8, 256, 0, stream>>>(bif0, bhf0, bsf0, G4);
    k_bias<<<8, 256, 0, stream>>>(bir0, bhr0, bsr0, G4);
    k_bias<<<8, 256, 0, stream>>>(bif1, bhf1, bsf1, G4);
    k_bias<<<8, 256, 0, stream>>>(bir1, bhr1, bsr1, G4);

    // 2) layer-0 input projections (WMMA GEMM)
    k_gemm_pre<<<8192, 256, 0, stream>>>(xbf, bwihf0, bsf0, pre0f, DD);
    k_gemm_pre<<<8192, 256, 0, stream>>>(xbf, bwihr0, bsr0, pre0r, DD);

    // 3) layer-0 recurrences (persistent, 8 WGs, global step barrier)
    k_zero_u32<<<1, 64, 0, stream>>>(cnt);
    k_lstm_rec<<<8, 256, 0, stream>>>(pre0f, bwhhf0, hst, h01, nullptr, cnt, 0, 0);
    k_zero_u32<<<1, 64, 0, stream>>>(cnt);
    k_lstm_rec<<<8, 256, 0, stream>>>(pre0r, bwhhr0, hst, h01, nullptr, cnt, 1, HH);

    // 4) layer-1 input projections (K = 1024, reads bf16 h01)
    k_gemm_pre<<<8192, 256, 0, stream>>>(h01, bwihf1, bsf1, pre1f, 2 * HH);
    k_gemm_pre<<<8192, 256, 0, stream>>>(h01, bwihr1, bsr1, pre1r, 2 * HH);

    // 5) layer-1 recurrences, writing f32 output directly
    k_zero_u32<<<1, 64, 0, stream>>>(cnt);
    k_lstm_rec<<<8, 256, 0, stream>>>(pre1f, bwhhf1, hst, nullptr, out, cnt, 0, 0);
    k_zero_u32<<<1, 64, 0, stream>>>(cnt);
    k_lstm_rec<<<8, 256, 0, stream>>>(pre1r, bwhhr1, hst, nullptr, out, cnt, 1, HH);
}
